// Llama4TextExperts_11020886082286
// MI455X (gfx1250) — matrix operand order repaired and checked by reading, not compiled, not run on hardware
//
#include <hip/hip_runtime.h>

// MoE (Llama4 experts) forward: x @ Wgu -> SwiGLU -> @ Wdn
// fp32 in HBM. Pipeline: K0 pre-converts x to bf16 (ws), K1/K2 are LDS-tiled
// bf16 WMMA GEMMs. Software pipeline per K-tile:
//   - A tiles (bf16): async global->LDS DMA, double-buffered LDS
//   - B tiles (fp32 weights): register-staged global loads for tile i+1
//     overlapped with WMMA of tile i, then v_cvt_pk_bf16_f32 + ds_store_b64.

typedef __bf16 bf16_t;
typedef __attribute__((ext_vector_type(16))) __bf16 v16bf;
typedef __attribute__((ext_vector_type(8)))  __bf16 v8bf;
typedef __attribute__((ext_vector_type(4)))  __bf16 v4bf;
typedef __attribute__((ext_vector_type(2)))  __bf16 v2bf;
typedef __attribute__((ext_vector_type(8)))  float  v8f;
typedef __attribute__((ext_vector_type(4)))  float  v4f;
typedef __attribute__((ext_vector_type(4)))  int    v4i;

#define AS1 __attribute__((address_space(1)))
#define AS3 __attribute__((address_space(3)))

#define NUM_E 8
#define TOK   1024
#define HID   2048
#define FEXP  4096
#define LDA   72      // LDS row stride in bf16 (144 B: 16B-aligned, breaks bank stride)

#if __has_builtin(__builtin_amdgcn_global_load_async_to_lds_b128) && \
    __has_builtin(__builtin_amdgcn_s_wait_asynccnt)
#define USE_ASYNC 1
#else
#define USE_ASYNC 0
#endif

__device__ __forceinline__ bf16_t f2bf(float f) {
  unsigned u = __builtin_bit_cast(unsigned, f);
  u += 0x7fffu + ((u >> 16) & 1u);
  unsigned short h = (unsigned short)(u >> 16);
  return __builtin_bit_cast(bf16_t, h);
}

// 4x fp32 -> 4x bf16, packed hardware cvt when available
__device__ __forceinline__ v4bf cvt4(v4f v) {
#if __has_builtin(__builtin_amdgcn_cvt_pk_bf16_f32)
  v2bf lo = __builtin_amdgcn_cvt_pk_bf16_f32(v[0], v[1]);
  v2bf hi = __builtin_amdgcn_cvt_pk_bf16_f32(v[2], v[3]);
  return __builtin_shufflevector(lo, hi, 0, 1, 2, 3);
#else
  v4bf r; r[0] = f2bf(v[0]); r[1] = f2bf(v[1]); r[2] = f2bf(v[2]); r[3] = f2bf(v[3]);
  return r;
#endif
}

// 16-byte global -> LDS copy; async DMA (ASYNCcnt) when the builtin exists
__device__ __forceinline__ void copy16_g2l(const bf16_t* g, bf16_t* l) {
#if USE_ASYNC
  __builtin_amdgcn_global_load_async_to_lds_b128(
      (AS1 v4i*)g, (AS3 v4i*)l, 0, 0);
#else
  *(v8bf*)l = *(const v8bf*)g;
#endif
}

__device__ __forceinline__ void wait_async_all() {
#if USE_ASYNC
  __builtin_amdgcn_s_wait_asynccnt(0);
#endif
}

__device__ __forceinline__ v16bf frag_cat(const bf16_t* p0, const bf16_t* p1) {
  v8bf lo = *(const v8bf*)p0;   // ds_load_b128
  v8bf hi = *(const v8bf*)p1;   // ds_load_b128
  return __builtin_shufflevector(lo, hi, 0,1,2,3,4,5,6,7,8,9,10,11,12,13,14,15);
}

__device__ __forceinline__ v8f wmma_bf16(v16bf a, v16bf b, v8f c) {
  return __builtin_amdgcn_wmma_f32_16x16x32_bf16(false, a, false, b, (short)0, c, false, false);
}

// ---------------------------------------------------------------------------
// Kernel 0: x (E,T,H) fp32 -> bf16, 8 elements per thread
// ---------------------------------------------------------------------------
__global__ __launch_bounds__(256)
void moe_cvt_x(const float* __restrict__ x, bf16_t* __restrict__ xbf) {
  size_t i = ((size_t)blockIdx.x * 256 + threadIdx.x) * 8;
  v4f a = *(const v4f*)(x + i);
  v4f b = *(const v4f*)(x + i + 4);
  v4bf pa = cvt4(a), pb = cvt4(b);
  *(v8bf*)(xbf + i) = __builtin_shufflevector(pa, pb, 0, 1, 2, 3, 4, 5, 6, 7);
}

// ---------------------------------------------------------------------------
// Kernel 1: gate_up GEMM + SwiGLU.  Tile: 128 tokens x 64 inter-features.
// 8 waves; wave (w>>1, w&1) owns rows [.. +32) x features [.. +32), holding
// gate AND up accumulators so SwiGLU is applied in-register.
// ---------------------------------------------------------------------------
__global__ __launch_bounds__(256)
void moe_gemm1_swiglu(const bf16_t* __restrict__ xbf,  // (E, T, H) bf16
                      const float* __restrict__ wgu,   // (E, H, 2F) fp32
                      bf16_t* __restrict__ inter)      // (E, T, F) bf16 ws
{
  __shared__ __align__(16) bf16_t As[2][128][LDA]; // [buf][m][k], async-filled
  __shared__ __align__(16) bf16_t Bg[64][LDA];     // [n][k] gate (transposed)
  __shared__ __align__(16) bf16_t Bu[64][LDA];     // [n][k] up   (transposed)

  const int tid = threadIdx.x;
  const int BLK_N = FEXP / 64;                 // 64
  const int BLK_M = TOK / 128;                 // 8
  const int e  = blockIdx.x / (BLK_M * BLK_N);
  const int rb = blockIdx.x % (BLK_M * BLK_N);
  const int t0 = (rb / BLK_N) * 128;
  const int f0 = (rb % BLK_N) * 64;

  const int lane = tid & 31;
  const int col  = lane & 15;
  const int hi   = lane >> 4;
  const int wave = tid >> 5;
  const int mB   = (wave >> 1) * 32;
  const int nB   = (wave & 1) * 32;

  v8f accg[2][2] = {};
  v8f accu[2][2] = {};

  const bf16_t* xg  = xbf + ((size_t)e * TOK + t0) * HID;
  const float*  wgB = wgu + (size_t)e * HID * (2 * FEXP) + f0;

  v4f gReg[4], uReg[4];                        // B staging registers

  auto loadB = [&](int k0) {
    const float* wk = wgB + (size_t)k0 * (2 * FEXP);
    #pragma unroll
    for (int it = 0; it < 4; ++it) {
      int idx = tid + it * 256;
      int n   = idx & 63;
      int kr  = (idx >> 6) * 4;
      const float* pg = wk + (size_t)kr * (2 * FEXP) + n;
      const float* pu = pg + FEXP;
      gReg[it] = v4f{ pg[0], pg[2 * FEXP], pg[4 * FEXP], pg[6 * FEXP] };
      uReg[it] = v4f{ pu[0], pu[2 * FEXP], pu[4 * FEXP], pu[6 * FEXP] };
    }
  };
  auto storeB = [&]() {
    #pragma unroll
    for (int it = 0; it < 4; ++it) {
      int idx = tid + it * 256;
      int n   = idx & 63;
      int kr  = (idx >> 6) * 4;
      *(v4bf*)&Bg[n][kr] = cvt4(gReg[it]);
      *(v4bf*)&Bu[n][kr] = cvt4(uReg[it]);
    }
  };
  auto loadA = [&](int k0, int buf) {
    #pragma unroll
    for (int it = 0; it < 4; ++it) {
      int idx = tid + it * 256;
      int r   = idx >> 3;
      int c8  = (idx & 7) * 8;
      copy16_g2l(xg + (size_t)r * HID + k0 + c8, &As[buf][r][c8]);
    }
  };

  loadB(0);
  loadA(0, 0);

  for (int k0 = 0, i = 0; k0 < HID; k0 += 64, ++i) {
    const int buf = i & 1;
    storeB();                      // tile i weights -> LDS (Bs free since barrier)
    wait_async_all();              // tile i A DMA complete
    __syncthreads();               // tile i fully visible
    if (k0 + 64 < HID) {           // prefetch tile i+1 under the WMMAs below
      loadA(k0 + 64, buf ^ 1);
      loadB(k0 + 64);
    }

    #pragma unroll
    for (int kk = 0; kk < 64; kk += 32) {
      v16bf bg[2], bu[2];
      #pragma unroll
      for (int nt = 0; nt < 2; ++nt) {
        const bf16_t* pg = &Bg[nB + nt * 16 + col][kk + hi * 8];
        const bf16_t* pu = &Bu[nB + nt * 16 + col][kk + hi * 8];
        bg[nt] = frag_cat(pg, pg + 16);
        bu[nt] = frag_cat(pu, pu + 16);
      }
      #pragma unroll
      for (int mt = 0; mt < 2; ++mt) {
        const bf16_t* pa = &As[buf][mB + mt * 16 + col][kk + hi * 8];
        v16bf a = frag_cat(pa, pa + 16);
        #pragma unroll
        for (int nt = 0; nt < 2; ++nt) {
          accg[mt][nt] = wmma_bf16(a, bg[nt], accg[mt][nt]);
          accu[mt][nt] = wmma_bf16(a, bu[nt], accu[mt][nt]);
        }
      }
    }
    __syncthreads();               // all reads of Bg/Bu done before next storeB
  }

  // --- SwiGLU in-register, write bf16 intermediate ---
  bf16_t* ig = inter + (size_t)e * TOK * FEXP;
  #pragma unroll
  for (int mt = 0; mt < 2; ++mt) {
    #pragma unroll
    for (int nt = 0; nt < 2; ++nt) {
      v8f g = accg[mt][nt];
      v8f u = accu[mt][nt];
      #pragma unroll
      for (int r = 0; r < 8; ++r) {
        float gv = g[r];
        float iv = u[r] * (gv / (1.0f + __expf(-gv)));   // up * silu(gate)
        int t = t0 + mB + mt * 16 + r + hi * 8;           // C-layout: M = r + 8*hi
        int f = f0 + nB + nt * 16 + col;                  // N = lane&15
        ig[(size_t)t * FEXP + f] = f2bf(iv);
      }
    }
  }
}

// ---------------------------------------------------------------------------
// Kernel 2: down GEMM.  Tile: 128 tokens x 128 hidden.  Wave = 32 x 64.
// ---------------------------------------------------------------------------
__global__ __launch_bounds__(256)
void moe_gemm2(const bf16_t* __restrict__ inter,  // (E, T, F) bf16
               const float* __restrict__ wdn,     // (E, F, H) fp32
               float* __restrict__ out)           // (E, T, H) fp32
{
  __shared__ __align__(16) bf16_t As[2][128][LDA]; // [buf][m][k], async-filled
  __shared__ __align__(16) bf16_t Bs[128][LDA];    // [n][k] (transposed)

  const int tid = threadIdx.x;
  const int BLK_N = HID / 128;                 // 16
  const int BLK_M = TOK / 128;                 // 8
  const int e  = blockIdx.x / (BLK_M * BLK_N);
  const int rb = blockIdx.x % (BLK_M * BLK_N);
  const int t0 = (rb / BLK_N) * 128;
  const int h0 = (rb % BLK_N) * 128;

  const int lane = tid & 31;
  const int col  = lane & 15;
  const int hi   = lane >> 4;
  const int wave = tid >> 5;
  const int mB   = (wave >> 1) * 32;
  const int nB   = (wave & 1) * 64;

  v8f acc[2][4] = {};

  const bf16_t* ag = inter + ((size_t)e * TOK + t0) * FEXP;
  const float*  dB = wdn + (size_t)e * FEXP * HID + h0;

  v4f bReg[8];                                 // B staging registers

  auto loadB = [&](int k0) {
    const float* wk = dB + (size_t)k0 * HID;
    #pragma unroll
    for (int it = 0; it < 8; ++it) {
      int idx = tid + it * 256;
      int n   = idx & 127;
      int kr  = (idx >> 7) * 4;
      const float* pb = wk + (size_t)kr * HID + n;
      bReg[it] = v4f{ pb[0], pb[HID], pb[2 * HID], pb[3 * HID] };
    }
  };
  auto storeB = [&]() {
    #pragma unroll
    for (int it = 0; it < 8; ++it) {
      int idx = tid + it * 256;
      int n   = idx & 127;
      int kr  = (idx >> 7) * 4;
      *(v4bf*)&Bs[n][kr] = cvt4(bReg[it]);
    }
  };
  auto loadA = [&](int k0, int buf) {
    #pragma unroll
    for (int it = 0; it < 4; ++it) {
      int idx = tid + it * 256;
      int r   = idx >> 3;
      int c8  = (idx & 7) * 8;
      copy16_g2l(ag + (size_t)r * FEXP + k0 + c8, &As[buf][r][c8]);
    }
  };

  loadB(0);
  loadA(0, 0);

  for (int k0 = 0, i = 0; k0 < FEXP; k0 += 64, ++i) {
    const int buf = i & 1;
    storeB();
    wait_async_all();
    __syncthreads();
    if (k0 + 64 < FEXP) {
      loadA(k0 + 64, buf ^ 1);
      loadB(k0 + 64);
    }

    #pragma unroll
    for (int kk = 0; kk < 64; kk += 32) {
      v16bf b[4];
      #pragma unroll
      for (int nt = 0; nt < 4; ++nt) {
        const bf16_t* pb = &Bs[nB + nt * 16 + col][kk + hi * 8];
        b[nt] = frag_cat(pb, pb + 16);
      }
      #pragma unroll
      for (int mt = 0; mt < 2; ++mt) {
        const bf16_t* pa = &As[buf][mB + mt * 16 + col][kk + hi * 8];
        v16bf a = frag_cat(pa, pa + 16);
        #pragma unroll
        for (int nt = 0; nt < 4; ++nt)
          acc[mt][nt] = wmma_bf16(a, b[nt], acc[mt][nt]);
      }
    }
    __syncthreads();
  }

  float* og = out + (size_t)e * TOK * HID;
  #pragma unroll
  for (int mt = 0; mt < 2; ++mt) {
    #pragma unroll
    for (int nt = 0; nt < 4; ++nt) {
      v8f a = acc[mt][nt];
      #pragma unroll
      for (int r = 0; r < 8; ++r) {
        int t = t0 + mB + mt * 16 + r + hi * 8;
        int h = h0 + nB + nt * 16 + col;
        og[(size_t)t * HID + h] = a[r];
      }
    }
  }
}

extern "C" void kernel_launch(void* const* d_in, const int* in_sizes, int n_in,
                              void* d_out, int out_size, void* d_ws, size_t ws_size,
                              hipStream_t stream) {
  const float* x   = (const float*)d_in[0];   // hidden_states (8192, 2048)
  const float* wgu = (const float*)d_in[1];   // gate_up_proj  (8, 2048, 8192)
  const float* wdn = (const float*)d_in[2];   // down_proj     (8, 4096, 2048)
  float* out = (float*)d_out;                 // (8192, 2048) fp32

  bf16_t* inter = (bf16_t*)d_ws;                                   // 64 MB
  bf16_t* xbf   = inter + (size_t)NUM_E * TOK * FEXP;              // +32 MB

  dim3 blk(256);
  // K0: 16M elems / 8 per thread / 256 = 8192 blocks
  moe_cvt_x<<<dim3((NUM_E * TOK * HID) / (8 * 256)), blk, 0, stream>>>(x, xbf);
  // K1: E * (T/128) * (F/64) = 4096 blocks
  moe_gemm1_swiglu<<<dim3(NUM_E * (TOK / 128) * (FEXP / 64)), blk, 0, stream>>>(xbf, wgu, inter);
  // K2: E * (T/128) * (H/128) = 1024 blocks
  moe_gemm2<<<dim3(NUM_E * (TOK / 128) * (HID / 128)), blk, 0, stream>>>(inter, wdn, out);
}
